// GPT_74929999446742
// MI455X (gfx1250) — compile-verified, matrix operand order checked by reading, not run on hardware
//
#include <hip/hip_runtime.h>
#include <cstdint>

typedef __attribute__((ext_vector_type(16))) _Float16 v16h;
typedef __attribute__((ext_vector_type(8)))  _Float16 v8h;
typedef __attribute__((ext_vector_type(4)))  _Float16 v4h;
typedef __attribute__((ext_vector_type(8)))  float    v8f;
typedef __attribute__((ext_vector_type(4)))  int      v4i;

#define Dm   768
#define Vv   32000
#define Ss   1024
#define Bb   4
#define Hh   12
#define HSs  64
#define TOK  (Bb * Ss)     // 4096
#define NL   12

#define AS1 __attribute__((address_space(1)))
#define AS3 __attribute__((address_space(3)))

// CDNA5 async global->LDS path (ASYNCcnt), with safe fallback.
#if defined(__gfx1250__) && __has_builtin(__builtin_amdgcn_global_load_async_to_lds_b128) && __has_builtin(__builtin_amdgcn_s_wait_asynccnt)
#define HAVE_ASYNC_LDS 1
#else
#define HAVE_ASYNC_LDS 0
#endif

// Copy 16 bytes (8 halves) global -> LDS.
__device__ __forceinline__ void copy16(const _Float16* g, _Float16* l) {
#if HAVE_ASYNC_LDS
    __builtin_amdgcn_global_load_async_to_lds_b128(
        (AS1 v4i*)(uintptr_t)g, (AS3 v4i*)(uintptr_t)l, 0, 0);
#else
    *(uint4*)l = *(const uint4*)g;
#endif
}

// Build a 16-half WMMA fragment from the two contiguous 8-half runs of a
// lane's stripe: elements [0..7] = base[8*half ..], [8..15] = base[16+8*half ..].
__device__ __forceinline__ v16h frag16(const _Float16* base, int half) {
    v8h lo = *(const v8h*)(base + 8 * half);
    v8h hi = *(const v8h*)(base + 16 + 8 * half);
    return __builtin_shufflevector(lo, hi, 0, 1, 2, 3, 4, 5, 6, 7,
                                           8, 9, 10, 11, 12, 13, 14, 15);
}

__device__ __forceinline__ float gelu_f(float x) {
    return 0.5f * x * (1.0f + erff(x * 0.7071067811865475f));
}

// ---------------------------------------------------------------------------
// f32 -> f16 conversion, 8 elements/thread, 128-bit accesses.
// ---------------------------------------------------------------------------
__global__ __launch_bounds__(256) void cvt_f32_f16(
    const float* __restrict__ src, _Float16* __restrict__ dst, long long n)
{
    long long i = ((long long)blockIdx.x * 256 + threadIdx.x) * 8;
    if (i + 8 <= n) {
        float4 a = *(const float4*)(src + i);
        float4 b = *(const float4*)(src + i + 4);
        v8h h;
        h[0] = (_Float16)a.x; h[1] = (_Float16)a.y;
        h[2] = (_Float16)a.z; h[3] = (_Float16)a.w;
        h[4] = (_Float16)b.x; h[5] = (_Float16)b.y;
        h[6] = (_Float16)b.z; h[7] = (_Float16)b.w;
        *(v8h*)(dst + i) = h;
    }
}

// ---------------------------------------------------------------------------
// f32 [K,N] -> f16 [N,K] transpose-convert, 64x64 tiles via LDS.
// Batched over blockIdx.z (stride K*N). K, N multiples of 64.
// ---------------------------------------------------------------------------
__global__ __launch_bounds__(256) void cvt_t(
    const float* __restrict__ src, _Float16* __restrict__ dst, int K, int N)
{
    __shared__ _Float16 t[64][65];
    const long long bofs = (long long)blockIdx.z * K * N;
    src += bofs; dst += bofs;
    const int nb = blockIdx.x * 64;
    const int kb = blockIdx.y * 64;
    const int tid = threadIdx.x;

#pragma unroll
    for (int i = 0; i < 4; ++i) {                 // 1024 float4 chunks
        int c = tid + i * 256;
        int row = c >> 4, n4 = (c & 15) * 4;      // k-row, 4 n's
        float4 v = *(const float4*)(src + (long long)(kb + row) * N + nb + n4);
        t[row][n4]     = (_Float16)v.x;
        t[row][n4 + 1] = (_Float16)v.y;
        t[row][n4 + 2] = (_Float16)v.z;
        t[row][n4 + 3] = (_Float16)v.w;
    }
    __syncthreads();
#pragma unroll
    for (int i = 0; i < 4; ++i) {                 // write [N,K], 4 k's per chunk
        int c = tid + i * 256;
        int n = c >> 4, k4 = (c & 15) * 4;
        v4h h;
        h[0] = t[k4][n]; h[1] = t[k4 + 1][n];
        h[2] = t[k4 + 2][n]; h[3] = t[k4 + 3][n];
        *(v4h*)(dst + (long long)(nb + n) * K + kb + k4) = h;
    }
}

// ---------------------------------------------------------------------------
// WMMA GEMM, f16 operands, f32 accumulate, double-buffered async-LDS tiles.
//   A [M,K] f16 (lda); B [N,K] f16 (ldb) -- always K-major (pre-transposed).
//   Outputs (optional each): C f32 [M,N], C16 f16 [M,N], C16T f16 [N,M]
//     value = resid + alpha * act(A*B^T + bias)
//   Batched over blockIdx.z (A shared; strides sB/sbias/sC).
//   act: 0 = none, 1 = exact GELU. All dims multiples of tile sizes.
// ---------------------------------------------------------------------------
#define BM 128
#define BN 64
#define BK 32

__global__ __launch_bounds__(256) void gemm_wmma(
    const _Float16* __restrict__ A, const _Float16* __restrict__ B,
    const float* __restrict__ bias, const float* __restrict__ resid,
    const float* __restrict__ alpha_ptr,
    float* __restrict__ C, _Float16* __restrict__ C16, _Float16* __restrict__ C16T,
    int M, int N, int K, int lda, int ldb, int ldc,
    long long sB, long long sbias, long long sC, int act)
{
    __shared__ __attribute__((aligned(16))) _Float16 As[2][BM][BK];
    __shared__ __attribute__((aligned(16))) _Float16 Bs[2][BN][BK];

    const int bz = blockIdx.z;
    const _Float16* Bb_ = B + (long long)bz * sB;
    const float* biasb = bias ? bias + (long long)bz * sbias : nullptr;
    float*     Cb   = C    ? C    + (long long)bz * sC : nullptr;
    _Float16*  C16b = C16  ? C16  + (long long)bz * sC : nullptr;
    _Float16*  C16Tb = C16T ? C16T + (long long)bz * sC : nullptr;

    const int m0 = blockIdx.y * BM;
    const int n0 = blockIdx.x * BN;
    const int tid  = threadIdx.x;
    const int wave = tid >> 5;
    const int lane = tid & 31;
    const int r    = lane & 15;
    const int half = lane >> 4;
    const int wm = (wave & 3) * 32;
    const int wn = (wave >> 2) * 32;

    // Stage one BK-slice into LDS buffer `buf` (3 async ops / wave / stage).
    auto stage = [&](int buf, int k0) {
        // A tile: 128x32 halves = 512 x 16B chunks (2/thread)
#pragma unroll
        for (int i = 0; i < 2; ++i) {
            int c = tid + i * 256;
            int row = c >> 2, k8 = (c & 3) * 8;
            copy16(&A[(long long)(m0 + row) * lda + k0 + k8], &As[buf][row][k8]);
        }
        // B tile: 64x32 halves = 256 x 16B chunks (1/thread), rows along K
        {
            int n = tid >> 2, k8 = (tid & 3) * 8;
            copy16(&Bb_[(long long)(n0 + n) * ldb + k0 + k8], &Bs[buf][n][k8]);
        }
    };

    v8f acc[2][2] = {};

    stage(0, 0);
    int buf = 0;
    for (int k0 = 0; k0 < K; k0 += BK, buf ^= 1) {
        const int kn = k0 + BK;
        if (kn < K) stage(buf ^ 1, kn);       // prefetch next slice
#if HAVE_ASYNC_LDS
        if (kn < K) __builtin_amdgcn_s_wait_asynccnt(3);  // next stage in flight
        else        __builtin_amdgcn_s_wait_asynccnt(0);
#endif
        __syncthreads();

        v16h af[2], bf[2];
#pragma unroll
        for (int t = 0; t < 2; ++t) {
            af[t] = frag16(&As[buf][wm + 16 * t + r][0], half);   // 2 x ds b128
            bf[t] = frag16(&Bs[buf][wn + 16 * t + r][0], half);   // 2 x ds b128
        }
#pragma unroll
        for (int ti = 0; ti < 2; ++ti)
#pragma unroll
            for (int tj = 0; tj < 2; ++tj)
                acc[ti][tj] = __builtin_amdgcn_wmma_f32_16x16x32_f16(
                    false, af[ti], false, bf[tj], (short)0, acc[ti][tj], false, false);
        __syncthreads();   // all reads done before next stage overwrites
    }

    const float alpha = alpha_ptr ? alpha_ptr[0] : 1.0f;
#pragma unroll
    for (int ti = 0; ti < 2; ++ti) {
#pragma unroll
        for (int tj = 0; tj < 2; ++tj) {
#pragma unroll
            for (int i = 0; i < 8; ++i) {
                int row = m0 + wm + 16 * ti + i + 8 * half;
                int col = n0 + wn + 16 * tj + r;
                float vv = acc[ti][tj][i];
                if (biasb) vv += biasb[col];
                if (act == 1) vv = gelu_f(vv);
                vv *= alpha;
                if (resid) vv += resid[(long long)row * ldc + col];
                if (Cb)    Cb[(long long)row * ldc + col] = vv;
                if (C16b)  C16b[(long long)row * ldc + col] = (_Float16)vv;
                if (C16Tb) C16Tb[(long long)col * M + row] = (_Float16)vv;
            }
        }
    }
}

// ---------------------------------------------------------------------------
// Fused causal attention (flash-style, online softmax), f16 in / f16 out.
//   Q/K layout: [H, B*S, HS] f16.  V^T layout: [H, HS, B*S] f16.
//   O layout: [B*S, D] f16 (col = h*64+hs).
//   grid (S/128, H, B), 256 threads; wave w owns 16 query rows.
// ---------------------------------------------------------------------------
__global__ __launch_bounds__(256) void attn_fused(
    const _Float16* __restrict__ Q, const _Float16* __restrict__ Kg,
    const _Float16* __restrict__ VT, _Float16* __restrict__ Og)
{
    __shared__ __attribute__((aligned(16))) _Float16 Pbuf[8][16][32];

    const int h  = blockIdx.y;
    const int b  = blockIdx.z;
    const int wave = threadIdx.x >> 5;
    const int lane = threadIdx.x & 31;
    const int r    = lane & 15;
    const int half = lane >> 4;
    const int qb = blockIdx.x * 128 + wave * 16;

    const long long base = ((long long)h * Bb + b) * Ss * HSs;
    const _Float16* qg = Q + base;
    const _Float16* kg = Kg + base;
    const _Float16* vtg = VT + (long long)h * TOK * HSs + (long long)b * Ss; // + hs*TOK + key

    // Q A-fragments: rows qb..qb+15, two K=32 chunks, 2 x b128 each
    v16h aq[2];
#pragma unroll
    for (int c = 0; c < 2; ++c)
        aq[c] = frag16(&qg[(long long)(qb + r) * HSs + c * 32], half);

    float mrow[8], lrow[8];
    v8f o[4] = {};
#pragma unroll
    for (int i = 0; i < 8; ++i) { mrow[i] = -1e30f; lrow[i] = 0.0f; }

    const int kend = qb + 16;
    for (int kb = 0; kb < kend; kb += 32) {
        v8f s[2] = {};
#pragma unroll
        for (int t = 0; t < 2; ++t) {
#pragma unroll
            for (int c = 0; c < 2; ++c) {
                v16h bk = frag16(&kg[(long long)(kb + 16 * t + r) * HSs + c * 32], half);
                s[t] = __builtin_amdgcn_wmma_f32_16x16x32_f16(
                    false, aq[c], false, bk, (short)0, s[t], false, false);
            }
        }
#pragma unroll
        for (int t = 0; t < 2; ++t)
#pragma unroll
            for (int i = 0; i < 8; ++i) {
                int key = kb + 16 * t + r;
                int row = qb + i + 8 * half;
                float sv = s[t][i] * 0.125f;            // 1/sqrt(64)
                if (key > row) sv = -1e30f;
                s[t][i] = sv;
            }
        float tmax[8];
#pragma unroll
        for (int i = 0; i < 8; ++i) tmax[i] = fmaxf(s[0][i], s[1][i]);
#pragma unroll
        for (int w = 1; w < 16; w <<= 1)
#pragma unroll
            for (int i = 0; i < 8; ++i)
                tmax[i] = fmaxf(tmax[i], __shfl_xor(tmax[i], w, 32));

        float corr[8], rsum[8];
#pragma unroll
        for (int i = 0; i < 8; ++i) {
            float nm = fmaxf(mrow[i], tmax[i]);
            corr[i] = __expf(mrow[i] - nm);
            mrow[i] = nm;
            s[0][i] = __expf(s[0][i] - nm);
            s[1][i] = __expf(s[1][i] - nm);
            rsum[i] = s[0][i] + s[1][i];
        }
#pragma unroll
        for (int w = 1; w < 16; w <<= 1)
#pragma unroll
            for (int i = 0; i < 8; ++i)
                rsum[i] += __shfl_xor(rsum[i], w, 32);
#pragma unroll
        for (int i = 0; i < 8; ++i) lrow[i] = lrow[i] * corr[i] + rsum[i];
#pragma unroll
        for (int t = 0; t < 4; ++t)
#pragma unroll
            for (int i = 0; i < 8; ++i) o[t][i] *= corr[i];

        // P (16x32) -> per-wave LDS slice -> A-fragment (in-order DS)
#pragma unroll
        for (int t = 0; t < 2; ++t)
#pragma unroll
            for (int i = 0; i < 8; ++i)
                Pbuf[wave][i + 8 * half][16 * t + r] = (_Float16)s[t][i];
        v16h ap = frag16(&Pbuf[wave][r][0], half);     // 2 x ds_load_b128
#pragma unroll
        for (int t = 0; t < 4; ++t) {
            // V^T key-stripe is contiguous: 2 x global b128
            v16h bv = frag16(&vtg[(long long)(16 * t + r) * TOK + kb], half);
            o[t] = __builtin_amdgcn_wmma_f32_16x16x32_f16(
                false, ap, false, bv, (short)0, o[t], false, false);
        }
    }

#pragma unroll
    for (int t = 0; t < 4; ++t)
#pragma unroll
        for (int i = 0; i < 8; ++i) {
            int row = qb + i + 8 * half;
            int col = h * HSs + 16 * t + r;
            Og[(long long)(b * Ss + row) * Dm + col] = (_Float16)(o[t][i] / lrow[i]);
        }
}

// ---------------------------------------------------------------------------
__global__ __launch_bounds__(256) void embed_kernel(
    const int* __restrict__ idx, const float* __restrict__ tok,
    const float* __restrict__ pos, float* __restrict__ x)
{
    const int toki = blockIdx.x;            // b*S + s
    const int s = toki & (Ss - 1);
    const int t = idx[toki];
    for (int d = threadIdx.x * 4; d < Dm; d += 1024) {
        float4 te = *(const float4*)(tok + (long long)t * Dm + d);
        float4 pe = *(const float4*)(pos + s * Dm + d);
        float4 o;
        o.x = te.x + pe.x * 0.03608439182435161f;
        o.y = te.y + pe.y * 0.03608439182435161f;
        o.z = te.z + pe.z * 0.03608439182435161f;
        o.w = te.w + pe.w * 0.03608439182435161f;
        *(float4*)(x + (long long)toki * Dm + d) = o;
    }
}

// LayerNorm over D=768, one block per row, emits f16 operand.
__global__ __launch_bounds__(256) void ln_kernel(
    const float* __restrict__ x, const float* __restrict__ g,
    const float* __restrict__ bta, _Float16* __restrict__ out)
{
    __shared__ float red[256];
    const int row = blockIdx.x;
    const int tid = threadIdx.x;
    const bool act = tid < (Dm / 4);
    const float* xr = x + (long long)row * Dm;

    float4 v = {0.0f, 0.0f, 0.0f, 0.0f};
    if (act) v = *(const float4*)(xr + tid * 4);
    red[tid] = v.x + v.y + v.z + v.w;
    __syncthreads();
    for (int st = 128; st > 0; st >>= 1) {
        if (tid < st) red[tid] += red[tid + st];
        __syncthreads();
    }
    const float mu = red[0] * (1.0f / Dm);
    __syncthreads();

    float vs = 0.0f;
    if (act) {
        float dx = v.x - mu, dy = v.y - mu, dz = v.z - mu, dw = v.w - mu;
        vs = dx * dx + dy * dy + dz * dz + dw * dw;
    }
    red[tid] = vs; __syncthreads();
    for (int st = 128; st > 0; st >>= 1) {
        if (tid < st) red[tid] += red[tid + st];
        __syncthreads();
    }
    const float rinv = rsqrtf(red[0] * (1.0f / Dm) + 1e-5f);
    if (act) {
        float4 gg = *(const float4*)(g + tid * 4);
        float4 bb = *(const float4*)(bta + tid * 4);
        v4h hv;
        hv[0] = (_Float16)((v.x - mu) * rinv * gg.x + bb.x);
        hv[1] = (_Float16)((v.y - mu) * rinv * gg.y + bb.y);
        hv[2] = (_Float16)((v.z - mu) * rinv * gg.z + bb.z);
        hv[3] = (_Float16)((v.w - mu) * rinv * gg.w + bb.w);
        *(v4h*)(out + (long long)row * Dm + tid * 4) = hv;
    }
}

__global__ __launch_bounds__(256) void zero_kernel(float* p) { p[0] = 0.0f; }

__global__ __launch_bounds__(256) void loss_kernel(
    const float* __restrict__ logits, const int* __restrict__ targets,
    float* __restrict__ loss_out)
{
    __shared__ float red[256];
    const int row = blockIdx.x;
    const int tid = threadIdx.x;
    const float* lp = logits + (long long)row * Vv;

    float mx = -1e30f;
    for (int i = tid * 4; i < Vv; i += 1024) {
        float4 v = *(const float4*)(lp + i);
        mx = fmaxf(mx, fmaxf(fmaxf(v.x, v.y), fmaxf(v.z, v.w)));
    }
    red[tid] = mx; __syncthreads();
    for (int st = 128; st > 0; st >>= 1) {
        if (tid < st) red[tid] = fmaxf(red[tid], red[tid + st]);
        __syncthreads();
    }
    mx = red[0]; __syncthreads();

    float sum = 0.0f;
    for (int i = tid * 4; i < Vv; i += 1024) {
        float4 v = *(const float4*)(lp + i);
        sum += __expf(v.x - mx) + __expf(v.y - mx) + __expf(v.z - mx) + __expf(v.w - mx);
    }
    red[tid] = sum; __syncthreads();
    for (int st = 128; st > 0; st >>= 1) {
        if (tid < st) red[tid] += red[tid + st];
        __syncthreads();
    }
    if (tid == 0) {
        float lse = mx + __logf(red[0]);
        float lpt = lp[targets[row]] - lse;
        atomicAdd(loss_out, -lpt * (1.0f / TOK));
    }
}

// ---------------------------------------------------------------------------
extern "C" void kernel_launch(void* const* d_in, const int* in_sizes, int n_in,
                              void* d_out, int out_size, void* d_ws, size_t ws_size,
                              hipStream_t stream) {
    const int*   idx      = (const int*)  d_in[0];
    const int*   targets  = (const int*)  d_in[1];
    const float* tok_emb  = (const float*)d_in[2];
    const float* pos_emb  = (const float*)d_in[3];
    const float* Wq = (const float*)d_in[4];  const float* bq = (const float*)d_in[5];
    const float* Wk = (const float*)d_in[6];  const float* bk = (const float*)d_in[7];
    const float* Wv = (const float*)d_in[8];  const float* bv = (const float*)d_in[9];
    const float* Wo = (const float*)d_in[10]; const float* bo = (const float*)d_in[11];
    const float* ln1_g = (const float*)d_in[12]; const float* ln1_b = (const float*)d_in[13];
    const float* ln2_g = (const float*)d_in[14]; const float* ln2_b = (const float*)d_in[15];
    const float* W1 = (const float*)d_in[16]; const float* b1 = (const float*)d_in[17];
    const float* W2 = (const float*)d_in[18]; const float* b2 = (const float*)d_in[19];
    const float* W3 = (const float*)d_in[20]; const float* b3 = (const float*)d_in[21];
    const float* att_scale = (const float*)d_in[22];
    const float* ffd_scale = (const float*)d_in[23];
    const float* lnf_g = (const float*)d_in[24]; const float* lnf_b = (const float*)d_in[25];

    float* logits = (float*)d_out;                 // [TOK, V]
    float* loss   = logits + (long long)TOK * Vv;  // [1]

    // ---- workspace carve: f32 residual + f16 operand buffers ----
    float* x = (float*)d_ws;                                   // [TOK, D] f32
    _Float16* h16  = (_Float16*)(x + (long long)TOK * Dm);     // [TOK, D]
    _Float16* q16  = h16  + (long long)TOK * Dm;               // [H, TOK, HS]
    _Float16* k16  = q16  + (long long)TOK * Dm;
    _Float16* vT16 = k16  + (long long)TOK * Dm;               // [H, HS, TOK]
    _Float16* o16  = vT16 + (long long)TOK * Dm;               // [TOK, D]
    _Float16* f1_16 = o16  + (long long)TOK * Dm;              // [TOK, 4D]
    _Float16* f2_16 = f1_16 + (long long)TOK * 4 * Dm;         // [TOK, 3D]
    // per-layer K-major (transposed) f16 weight scratch
    _Float16* wq16 = f2_16 + (long long)TOK * 3 * Dm;          // [H, HS, D]
    _Float16* wk16 = wq16 + (long long)Hh * Dm * HSs;
    _Float16* wv16 = wk16 + (long long)Hh * Dm * HSs;
    _Float16* wo16 = wv16 + (long long)Hh * Dm * HSs;          // [D, D]
    _Float16* w1_16 = wo16 + (long long)Dm * Dm;               // [4D, D]
    _Float16* w2_16 = w1_16 + (long long)Dm * 4 * Dm;          // [3D, 4D]
    _Float16* w3_16 = w2_16 + (long long)4 * Dm * 3 * Dm;      // [D, 3D]
    _Float16* tok16 = w3_16 + (long long)3 * Dm * Dm;          // [V, D] (already K-major)

    auto cvt = [&](const float* s, _Float16* d, long long n) {
        cvt_f32_f16<<<(int)((n / 8 + 255) / 256), 256, 0, stream>>>(s, d, n);
    };
    auto cvtT = [&](const float* s, _Float16* d, int K, int N, int batch) {
        cvt_t<<<dim3(N / 64, K / 64, batch), 256, 0, stream>>>(s, d, K, N);
    };

    const dim3 blk(256);
    const long long szQKV = (long long)Hh * Dm * HSs;   // 589824

    embed_kernel<<<TOK, blk, 0, stream>>>(idx, tok_emb, pos_emb, x);
    cvt(tok_emb, tok16, (long long)Vv * Dm);            // tied head weights, once

    for (int l = 0; l < NL; ++l) {
        // per-layer weight transpose-convert f32 [K,N] -> f16 [N,K]
        cvtT(Wq + l * szQKV, wq16, Dm, HSs, Hh);
        cvtT(Wk + l * szQKV, wk16, Dm, HSs, Hh);
        cvtT(Wv + l * szQKV, wv16, Dm, HSs, Hh);
        cvtT(Wo + (long long)l * Dm * Dm, wo16, Dm, Dm, 1);
        cvtT(W1 + (long long)l * Dm * 4 * Dm, w1_16, Dm, 4 * Dm, 1);
        cvtT(W2 + (long long)l * 4 * Dm * 3 * Dm, w2_16, 4 * Dm, 3 * Dm, 1);
        cvtT(W3 + (long long)l * 3 * Dm * Dm, w3_16, 3 * Dm, Dm, 1);

        ln_kernel<<<TOK, blk, 0, stream>>>(x, ln1_g + l * Dm, ln1_b + l * Dm, h16);

        // QKV batched over heads (B = W^T [HS, D], ldb = D)
        dim3 gqkv(1, TOK / BM, Hh);
        gemm_wmma<<<gqkv, blk, 0, stream>>>(h16, wq16, bq + l * Hh * HSs,
            nullptr, nullptr, nullptr, q16, nullptr, TOK, HSs, Dm, Dm, Dm, HSs,
            (long long)HSs * Dm, HSs, (long long)TOK * HSs, 0);
        gemm_wmma<<<gqkv, blk, 0, stream>>>(h16, wk16, bk + l * Hh * HSs,
            nullptr, nullptr, nullptr, k16, nullptr, TOK, HSs, Dm, Dm, Dm, HSs,
            (long long)HSs * Dm, HSs, (long long)TOK * HSs, 0);
        // V: transposed epilogue -> V^T [H, HS, TOK]
        gemm_wmma<<<gqkv, blk, 0, stream>>>(h16, wv16, bv + l * Hh * HSs,
            nullptr, nullptr, nullptr, nullptr, vT16, TOK, HSs, Dm, Dm, Dm, HSs,
            (long long)HSs * Dm, HSs, (long long)TOK * HSs, 0);

        attn_fused<<<dim3(Ss / 128, Hh, Bb), blk, 0, stream>>>(q16, k16, vT16, o16);

        // x = x + att_scale[l] * (o @ Wo + bo)
        gemm_wmma<<<dim3(Dm / BN, TOK / BM, 1), blk, 0, stream>>>(
            o16, wo16, bo + l * Dm, x, att_scale + l, x, nullptr, nullptr,
            TOK, Dm, Dm, Dm, Dm, Dm, 0, 0, 0, 0);

        ln_kernel<<<TOK, blk, 0, stream>>>(x, ln2_g + l * Dm, ln2_b + l * Dm, h16);

        // f1 = gelu(h @ W1 + b1) -> f16 operand
        gemm_wmma<<<dim3(4 * Dm / BN, TOK / BM, 1), blk, 0, stream>>>(
            h16, w1_16, b1 + l * 4 * Dm, nullptr, nullptr, nullptr, f1_16, nullptr,
            TOK, 4 * Dm, Dm, Dm, Dm, 4 * Dm, 0, 0, 0, 1);
        // f2 = gelu(f1 @ W2 + b2) -> f16 operand
        gemm_wmma<<<dim3(3 * Dm / BN, TOK / BM, 1), blk, 0, stream>>>(
            f1_16, w2_16, b2 + l * 3 * Dm, nullptr, nullptr, nullptr, f2_16, nullptr,
            TOK, 3 * Dm, 4 * Dm, 4 * Dm, 4 * Dm, 3 * Dm, 0, 0, 0, 1);
        // x = x + ffd_scale[l] * (f2 @ W3 + b3)
        gemm_wmma<<<dim3(Dm / BN, TOK / BM, 1), blk, 0, stream>>>(
            f2_16, w3_16, b3 + l * Dm, x, ffd_scale + l, x, nullptr, nullptr,
            TOK, Dm, 3 * Dm, 3 * Dm, 3 * Dm, Dm, 0, 0, 0, 0);
    }

    ln_kernel<<<TOK, blk, 0, stream>>>(x, lnf_g, lnf_b, h16);

    // logits = h @ tok_emb^T (tok16 is [V, D], already K-major), f32 to d_out
    gemm_wmma<<<dim3(Vv / BN, TOK / BM, 1), blk, 0, stream>>>(
        h16, tok16, nullptr, nullptr, nullptr, logits, nullptr, nullptr,
        TOK, Vv, Dm, Dm, Dm, Vv, 0, 0, 0, 0);

    zero_kernel<<<1, 1, 0, stream>>>(loss);
    loss_kernel<<<TOK, blk, 0, stream>>>(logits, targets, loss);
}
